// Time_FeatureExtractor_attention_65867618451581
// MI455X (gfx1250) — matrix operand order repaired and checked by reading, not compiled, not run on hardware
//
#include <hip/hip_runtime.h>

// ---------------------------------------------------------------------------
// Problem constants (from reference)
// ---------------------------------------------------------------------------
#define Bsz   256
#define Lsz   168
#define Csz   36
#define Fsz   64
#define Dsz   256
#define STEP  56                 // L / 3
#define NPOS  (Bsz*Lsz*Csz)      // 1,548,288
#define KFC   (Fsz*Csz)          // 2304
#define NROW  (Bsz*STEP)         // 14336
#define PLANE (Lsz*Csz)          // 6048
#define LTILES 42                // L / 4 ; N-tile = 4 rows = 144 positions

typedef __attribute__((ext_vector_type(16))) __bf16        v16bf;
typedef __attribute__((ext_vector_type(8)))  float         v8f;
typedef __attribute__((ext_vector_type(4)))  unsigned int  u32x4;
typedef __attribute__((ext_vector_type(4)))  int           i32x4;
typedef __attribute__((ext_vector_type(8)))  int           i32x8;

__device__ __forceinline__ unsigned short f2bf(float f) {
    unsigned int u = __float_as_uint(f);
    u += 0x7FFFu + ((u >> 16) & 1u);          // round-to-nearest-even
    return (unsigned short)(u >> 16);
}

// ---------------------------------------------------------------------------
// fp32 -> bf16 flat converter (weights)
// ---------------------------------------------------------------------------
__global__ void f32_to_bf16_k(const float* __restrict__ in,
                              unsigned short* __restrict__ out, int n) {
    int i = blockIdx.x * 256 + threadIdx.x;
    if (i < n) out[i] = f2bf(in[i]);
}

// ---------------------------------------------------------------------------
// conv0: 1 -> 64 channels, 5-tap along L, SAME pad, ReLU, bf16 output
// ---------------------------------------------------------------------------
__global__ void conv0_k(const float* __restrict__ x,
                        const float* __restrict__ w0,
                        const float* __restrict__ b0,
                        unsigned short* __restrict__ hout) {
    __shared__ float w[Fsz * 5];
    __shared__ float bb[Fsz];
    for (int i = threadIdx.x; i < Fsz * 5; i += 256) w[i] = w0[i];
    for (int i = threadIdx.x; i < Fsz;     i += 256) bb[i] = b0[i];
    __syncthreads();

    int n = blockIdx.x * 256 + threadIdx.x;
    int b = n / PLANE;
    int r = n - b * PLANE;
    int l = r / Csz;
    int c = r - l * Csz;

    float xv[5];
#pragma unroll
    for (int t = 0; t < 5; ++t) {
        int ll = l + t - 2;
        xv[t] = (ll >= 0 && ll < Lsz) ? x[((size_t)b * Lsz + ll) * Csz + c] : 0.f;
    }
    size_t base = (size_t)b * (Fsz * PLANE) + (size_t)l * Csz + c;
    for (int f = 0; f < Fsz; ++f) {
        float s = bb[f];
#pragma unroll
        for (int t = 0; t < 5; ++t) s += w[f * 5 + t] * xv[t];
        s = fmaxf(s, 0.f);
        hout[base + (size_t)f * PLANE] = f2bf(s);
    }
}

// ---------------------------------------------------------------------------
// conv1/2/3 implicit GEMM with TDM-staged input slab.
//   block tile: M=64 out-channels x N=144 positions (= 4 L rows, never crosses b)
//   input slab h[b, 0..63, 4tb-2 .. 4tb+5, :] loaded once via tensor_load_to_lds
//   K = 320 (64 ch * 5 taps) in 10 chunks of 32; bf16 WMMA, f32 accum
// ---------------------------------------------------------------------------
__global__ void conv_wmma_k(const unsigned short* __restrict__ hin,   // bf16 (B,F,L,C)
                            const unsigned short* __restrict__ wbf,   // bf16 [64][320]
                            const float* __restrict__ bias,
                            unsigned short* __restrict__ hout) {      // bf16 (B,F,L,C)
    __shared__ __align__(16) unsigned short inT[Fsz * 288]; // <=8 rows * 36 per channel
    __shared__ __align__(16) unsigned int   Bl[144][20];    // B panel, 80B row stride

    const int tid  = threadIdx.x;
    const int wave = tid >> 5, lane = tid & 31;
    const int lh   = lane & 15;
    const int aq   = (lane < 16) ? 0 : 1;   // A frag quad select (contiguous K)
    const int bq   = (lane < 16) ? 0 : 2;   // B frag quad select (contiguous K)
    const int mrel = (lane < 16) ? 0 : 8;
    const int tb   = blockIdx.x;            // covers l = 4tb .. 4tb+3
    const int b    = blockIdx.y;

    const int l0   = 4 * tb;
    const int llo  = (l0 - 2 < 0) ? 0 : (l0 - 2);
    const int lhi  = (l0 + 5 > Lsz - 1) ? (Lsz - 1) : (l0 + 5);
    const int span = (lhi - llo + 1) * Csz;          // valid elems per channel
    const int xoff = (llo - (l0 - 2)) * Csz;         // 72 iff tb==0, else 0

    // --- one TDM load per block: 2-D tile (span x 64 channels) -> LDS ---
    if (wave == 0) {
        unsigned long long ga = (unsigned long long)(const void*)
            (hin + ((size_t)b * Fsz * Lsz + llo) * Csz);
        unsigned ldsOff = (unsigned)(unsigned long long)(const void*)&inT[0];
        u32x4 g0;
        g0[0] = 1u;                                          // count=1, user mode
        g0[1] = ldsOff;                                      // LDS byte address
        g0[2] = (unsigned)ga;                                // global addr lo
        g0[3] = ((unsigned)(ga >> 32) & 0x1FFFFFFu) | (2u << 30); // addr hi | type=2
        i32x8 g1;
        g1[0] = 0x00010000;                                  // data_size = 2 bytes
        g1[1] = (int)(((unsigned)span & 0xFFFFu) << 16);     // tensor_dim0 lo16
        g1[2] = (int)(((unsigned)span >> 16) | (64u << 16)); // td0 hi16 | tensor_dim1 lo16
        g1[3] = (int)((unsigned)span << 16);                 // td1 hi16(0) | tile_dim0
        g1[4] = 64;                                          // tile_dim1=64, tile_dim2=0
        g1[5] = PLANE;                                       // tensor_dim0_stride (elems)
        g1[6] = 0;
        g1[7] = 0;
        i32x4 z4 = {0, 0, 0, 0};
        i32x8 z8 = {0, 0, 0, 0, 0, 0, 0, 0};
        __builtin_amdgcn_tensor_load_to_lds(g0, g1, z4, z4, z8, 0);
        __builtin_amdgcn_s_wait_tensorcnt(0);
    }

    const int mrow = 16 * wave + lh;
    const u32x4* arow = (const u32x4*)(wbf + (size_t)mrow * 320);

    v8f acc[9] = {};
    for (int kc = 0; kc < 10; ++kc) {
        __syncthreads();
        // stage B panel from LDS slab: 144 cols x 16 dword k-pairs
        for (int it = 0; it < 18; ++it) {
            int idx = tid + it * 128;      // 0..2303
            int j = idx >> 4;              // column (position in tile)
            int m = idx & 15;              // k-pair dword
            unsigned int pack = 0;
#pragma unroll
            for (int hh = 0; hh < 2; ++hh) {
                int k  = kc * 32 + 2 * m + hh;
                int fi = k / 5;
                int t  = k - fi * 5;
                int xx = j + t * Csz - xoff;   // slab-relative offset
                unsigned short e = 0;
                if ((unsigned)xx < (unsigned)span) e = inT[fi * span + xx];
                pack |= (unsigned)e << (16 * hh);
            }
            Bl[j][m] = pack;
        }
        __syncthreads();

        union { u32x4 q[2]; v16bf v; } a;
        a.q[0] = arow[kc * 4 + 2 * aq];
        a.q[1] = arow[kc * 4 + 2 * aq + 1];
#pragma unroll
        for (int nt = 0; nt < 9; ++nt) {
            int col = nt * 16 + lh;
            union { u32x4 q[2]; v16bf v; } bb;
            const u32x4* br = (const u32x4*)&Bl[col][0];
            bb.q[0] = br[bq];
            bb.q[1] = br[bq + 1];
            acc[nt] = __builtin_amdgcn_wmma_f32_16x16x32_bf16(
                false, a.v, false, bb.v, (short)0, acc[nt], false, false);
        }
    }

    const size_t bbase = (size_t)b * (Fsz * PLANE) + (size_t)l0 * Csz;
#pragma unroll
    for (int nt = 0; nt < 9; ++nt) {
        int j = nt * 16 + lh;
#pragma unroll
        for (int vr = 0; vr < 8; ++vr) {
            int fo = 16 * wave + vr + mrel;
            float val = fmaxf(acc[nt][vr] + bias[fo], 0.f);
            hout[bbase + (size_t)fo * PLANE + j] = f2bf(val);
        }
    }
}

// ---------------------------------------------------------------------------
// attention: one block (128 thr) per (b,s)
// ---------------------------------------------------------------------------
__global__ void attn_k(const unsigned short* __restrict__ h3,    // bf16 (B,F,L,C)
                       const unsigned short* __restrict__ wp,    // bf16 4x[64][192]
                       const float* __restrict__ gammap,
                       unsigned short* __restrict__ obuf) {      // bf16 [NROW][2304]
    __shared__ __align__(16) unsigned int xcl[48][96];  // xc bf16 [c][k]; reused as sc f32
    __shared__ __align__(16) unsigned int fl[48][32];   // f bf16  [i=c][q]
    __shared__ __align__(16) unsigned int gl[48][32];   // g bf16  [j=c][q]
    __shared__ __align__(16) unsigned int hvl[64][32];  // hv bf16 [q][i], i padded to 64
    __shared__ float        v1l[64][48];
    __shared__ __align__(16) unsigned int bl[48][32];   // beta bf16 [j][i], i padded
    float (*sc)[48] = (float (*)[48]) & xcl[0][0];

    const int tid  = threadIdx.x;
    const int wave = tid >> 5, lane = tid & 31;
    const int lh   = lane & 15;
    const int aq   = (lane < 16) ? 0 : 1;
    const int bq   = (lane < 16) ? 0 : 2;
    const int mrel = (lane < 16) ? 0 : 8;
    const int bs   = blockIdx.x;
    const int b    = bs / STEP;
    const int s    = bs - b * STEP;
    const float gamma = gammap[0];

    for (int i = tid; i < 48 * 32; i += 128) {
        fl[i >> 5][i & 31] = 0u; gl[i >> 5][i & 31] = 0u; bl[i >> 5][i & 31] = 0u;
    }
    for (int i = tid; i < 64 * 32; i += 128) hvl[i >> 5][i & 31] = 0u;
    for (int i = tid; i < 48 * 96; i += 128) xcl[i / 96][i % 96] = 0u;
    __syncthreads();

    for (int idx = tid; idx < 192 * Csz; idx += 128) {
        int k = idx / Csz;
        int c = idx - k * Csz;
        int f = k / 3, d = k - f * 3;
        unsigned short v = h3[(((size_t)b * Fsz + f) * Lsz + (s * 3 + d)) * Csz + c];
        ((unsigned short*)xcl)[c * 192 + k] = v;
    }
    __syncthreads();

    // --- projections q/k/v/v1: M=64 (wave=16 rows), N=48, K=192 ---
    for (int wm = 0; wm < 4; ++wm) {
        v8f acc[3] = {};
        const int mrow = 16 * wave + lh;
        const u32x4* arow = (const u32x4*)(wp + (size_t)wm * Fsz * 192 + (size_t)mrow * 192);
        for (int kc = 0; kc < 6; ++kc) {
            union { u32x4 q[2]; v16bf v; } a;
            a.q[0] = arow[kc * 4 + 2 * aq];
            a.q[1] = arow[kc * 4 + 2 * aq + 1];
#pragma unroll
            for (int nt = 0; nt < 3; ++nt) {
                int col = nt * 16 + lh;
                union { u32x4 q[2]; v16bf v; } bb;
                const u32x4* br = (const u32x4*)&xcl[col][kc * 16];
                bb.q[0] = br[bq == 0 ? 0 : 2];
                bb.q[1] = br[bq == 0 ? 1 : 3];
                acc[nt] = __builtin_amdgcn_wmma_f32_16x16x32_bf16(
                    false, a.v, false, bb.v, (short)0, acc[nt], false, false);
            }
        }
#pragma unroll
        for (int nt = 0; nt < 3; ++nt)
#pragma unroll
            for (int vr = 0; vr < 8; ++vr) {
                int q = 16 * wave + vr + mrel;
                int c = nt * 16 + lh;
                float val = acc[nt][vr];
                if (wm == 3) {
                    v1l[q][c] = val;
                } else if (c < Csz) {
                    unsigned short e = f2bf(val);
                    if      (wm == 0) ((unsigned short*)fl)[c * 64 + q] = e;
                    else if (wm == 1) ((unsigned short*)gl)[c * 64 + q] = e;
                    else              ((unsigned short*)hvl)[q * 64 + c] = e;
                }
            }
        __syncthreads();
    }

    // --- scores[i,j] = sum_q f[q,i] g[q,j]; waves 0..2, i-tile = wave ---
    if (wave < 3) {
        v8f acc[3] = {};
        const int irow = 16 * wave + lh;
        for (int kc = 0; kc < 2; ++kc) {
            union { u32x4 q[2]; v16bf v; } a;
            const u32x4* ar = (const u32x4*)&fl[irow][kc * 16];
            a.q[0] = ar[2 * aq];
            a.q[1] = ar[2 * aq + 1];
#pragma unroll
            for (int nt = 0; nt < 3; ++nt) {
                int col = nt * 16 + lh;
                union { u32x4 q[2]; v16bf v; } bb;
                const u32x4* br = (const u32x4*)&gl[col][kc * 16];
                bb.q[0] = br[bq == 0 ? 0 : 2];
                bb.q[1] = br[bq == 0 ? 1 : 3];
                acc[nt] = __builtin_amdgcn_wmma_f32_16x16x32_bf16(
                    false, a.v, false, bb.v, (short)0, acc[nt], false, false);
            }
        }
#pragma unroll
        for (int nt = 0; nt < 3; ++nt)
#pragma unroll
            for (int vr = 0; vr < 8; ++vr) {
                int i = 16 * wave + vr + mrel;
                int j = nt * 16 + lh;
                sc[j][i] = acc[nt][vr];
            }
    }
    __syncthreads();

    // --- softmax over i per column j ---
    if (tid < Csz) {
        int j = tid;
        float m = sc[j][0];
        for (int i = 1; i < Csz; ++i) m = fmaxf(m, sc[j][i]);
        float sum = 0.f;
        for (int i = 0; i < Csz; ++i) sum += __expf(sc[j][i] - m);
        float inv = 1.f / sum;
        for (int i = 0; i < Csz; ++i)
            ((unsigned short*)bl)[j * 64 + i] = f2bf(__expf(sc[j][i] - m) * inv);
    }
    __syncthreads();

    // --- o[q,j] = relu( gamma * sum_i hv[q,i] beta[i,j] + v1[q,j] ) ---
    {
        v8f acc[3] = {};
        const int qrow = 16 * wave + lh;
        for (int kc = 0; kc < 2; ++kc) {
            union { u32x4 q[2]; v16bf v; } a;
            const u32x4* ar = (const u32x4*)&hvl[qrow][kc * 16];
            a.q[0] = ar[2 * aq];
            a.q[1] = ar[2 * aq + 1];
#pragma unroll
            for (int nt = 0; nt < 3; ++nt) {
                int col = nt * 16 + lh;
                union { u32x4 q[2]; v16bf v; } bb;
                const u32x4* br = (const u32x4*)&bl[col][kc * 16];
                bb.q[0] = br[bq == 0 ? 0 : 2];
                bb.q[1] = br[bq == 0 ? 1 : 3];
                acc[nt] = __builtin_amdgcn_wmma_f32_16x16x32_bf16(
                    false, a.v, false, bb.v, (short)0, acc[nt], false, false);
            }
        }
#pragma unroll
        for (int nt = 0; nt < 3; ++nt)
#pragma unroll
            for (int vr = 0; vr < 8; ++vr) {
                int q = 16 * wave + vr + mrel;
                int j = nt * 16 + lh;
                float val = fmaxf(gamma * acc[nt][vr] + v1l[q][j], 0.f);
                if (j < Csz)
                    obuf[(size_t)bs * KFC + q * Csz + j] = f2bf(val);
            }
    }
}

// ---------------------------------------------------------------------------
// FC: out[row,d] = relu( sum_k o[row,k]*fc_w[d,k] + fc_b[d] )
// ---------------------------------------------------------------------------
__global__ void fc_k(const unsigned short* __restrict__ A,   // bf16 [NROW][2304]
                     const unsigned short* __restrict__ W,   // bf16 [256][2304]
                     const float* __restrict__ bias,
                     float* __restrict__ out) {
    const int tid  = threadIdx.x;
    const int wave = tid >> 5, lane = tid & 31;
    const int lh   = lane & 15;
    const int aq   = (lane < 16) ? 0 : 1;
    const int bq   = (lane < 16) ? 0 : 2;
    const int mrel = (lane < 16) ? 0 : 8;
    const int rbase = blockIdx.x * 64 + 16 * wave;
    const int cbase = blockIdx.y * 64;

    const u32x4* arow = (const u32x4*)(A + (size_t)(rbase + lh) * KFC);
    const u32x4* brow[4];
#pragma unroll
    for (int nt = 0; nt < 4; ++nt)
        brow[nt] = (const u32x4*)(W + (size_t)(cbase + nt * 16 + lh) * KFC);

    v8f acc[4] = {};
    for (int kc = 0; kc < 72; ++kc) {
        __builtin_prefetch((const char*)(arow + (kc + 8) * 4), 0, 1);
        union { u32x4 q[2]; v16bf v; } a;
        a.q[0] = arow[kc * 4 + 2 * aq];
        a.q[1] = arow[kc * 4 + 2 * aq + 1];
#pragma unroll
        for (int nt = 0; nt < 4; ++nt) {
            union { u32x4 q[2]; v16bf v; } bb;
            bb.q[0] = brow[nt][kc * 4 + bq];
            bb.q[1] = brow[nt][kc * 4 + bq + 1];
            acc[nt] = __builtin_amdgcn_wmma_f32_16x16x32_bf16(
                false, a.v, false, bb.v, (short)0, acc[nt], false, false);
        }
    }
#pragma unroll
    for (int nt = 0; nt < 4; ++nt)
#pragma unroll
        for (int vr = 0; vr < 8; ++vr) {
            int row = rbase + vr + mrel;
            int col = cbase + nt * 16 + lh;
            out[(size_t)row * Dsz + col] = fmaxf(acc[nt][vr] + bias[col], 0.f);
        }
}

// ---------------------------------------------------------------------------
// launch
// ---------------------------------------------------------------------------
extern "C" void kernel_launch(void* const* d_in, const int* in_sizes, int n_in,
                              void* d_out, int out_size, void* d_ws, size_t ws_size,
                              hipStream_t stream) {
    const float* x     = (const float*)d_in[0];
    const float* w0    = (const float*)d_in[1];
    const float* b0    = (const float*)d_in[2];
    const float* w1    = (const float*)d_in[3];
    const float* b1    = (const float*)d_in[4];
    const float* w2    = (const float*)d_in[5];
    const float* b2    = (const float*)d_in[6];
    const float* w3    = (const float*)d_in[7];
    const float* b3    = (const float*)d_in[8];
    const float* wq    = (const float*)d_in[9];
    const float* wk    = (const float*)d_in[10];
    const float* wv    = (const float*)d_in[11];
    const float* wv1   = (const float*)d_in[12];
    const float* gamma = (const float*)d_in[13];
    const float* fcw   = (const float*)d_in[14];
    const float* fcb   = (const float*)d_in[15];
    float* out = (float*)d_out;

    const size_t hElems = (size_t)Bsz * Fsz * Lsz * Csz;    // 99,090,432
    unsigned short* hA  = (unsigned short*)d_ws;
    unsigned short* hB  = hA + hElems;
    unsigned short* wc  = hB + hElems;          // conv weights bf16: 3 * 20480
    unsigned short* wpj = wc + 3 * 20480;       // proj weights bf16: 4 * 12288
    unsigned short* fw  = wpj + 4 * 12288;      // fc_w bf16: 589824
    unsigned short* obuf = hA;                  // reuse hA after conv3

    f32_to_bf16_k<<<(20480 + 255) / 256, 256, 0, stream>>>(w1, wc + 0,      20480);
    f32_to_bf16_k<<<(20480 + 255) / 256, 256, 0, stream>>>(w2, wc + 20480,  20480);
    f32_to_bf16_k<<<(20480 + 255) / 256, 256, 0, stream>>>(w3, wc + 40960,  20480);
    f32_to_bf16_k<<<(12288 + 255) / 256, 256, 0, stream>>>(wq,  wpj + 0,     12288);
    f32_to_bf16_k<<<(12288 + 255) / 256, 256, 0, stream>>>(wk,  wpj + 12288, 12288);
    f32_to_bf16_k<<<(12288 + 255) / 256, 256, 0, stream>>>(wv,  wpj + 24576, 12288);
    f32_to_bf16_k<<<(12288 + 255) / 256, 256, 0, stream>>>(wv1, wpj + 36864, 12288);
    f32_to_bf16_k<<<(589824 + 255) / 256, 256, 0, stream>>>(fcw, fw, 589824);

    conv0_k<<<NPOS / 256, 256, 0, stream>>>(x, w0, b0, hA);
    conv_wmma_k<<<dim3(LTILES, Bsz), 128, 0, stream>>>(hA, wc + 0,     b1, hB);
    conv_wmma_k<<<dim3(LTILES, Bsz), 128, 0, stream>>>(hB, wc + 20480, b2, hA);
    conv_wmma_k<<<dim3(LTILES, Bsz), 128, 0, stream>>>(hA, wc + 40960, b3, hB);

    attn_k<<<NROW, 128, 0, stream>>>(hB, wpj, gamma, obuf);

    fc_k<<<dim3(NROW / 64, Dsz / 64), 128, 0, stream>>>(obuf, fw, fcb, out);
}
